// RAINCOAT_57501022159587
// MI455X (gfx1250) — compile-verified, matrix operand order checked by reading, not run on hardware
//
#include <hip/hip_runtime.h>
#include <hip/hip_bf16.h>

// ---------------------------------------------------------------------------
// Problem constants (from the reference)
// ---------------------------------------------------------------------------
#define BATCH 128
#define CIN   32
#define LSEQ  2048
#define MIDC  128
#define FMC   128
#define NCLS  10

#define NT_PER_BLOCK 4              // N tiles (of 16) per wave
#define NBLK (LSEQ / 64)            // 32 N-blocks per batch row
#define NPART (BATCH * NBLK)        // stats partials per channel (4096)

typedef __attribute__((ext_vector_type(16))) __bf16 v16bf;
typedef __attribute__((ext_vector_type(8)))  __bf16 v8bf;
typedef __attribute__((ext_vector_type(8)))  float  v8f;

// ---------------------------------------------------------------------------
// WMMA GEMM:  Yraw[b,o,l] = sum_{tap,k} W[tap][o][k] * X[b][k][l + tap - (TAPS>>1)]
//
// Block = 256 threads = 8 waves.  Per block: M=128 x N=64 outputs of one batch.
// Each wave owns one 16-row M tile and 4 N tiles, so the A fragment (two 16B
// global bf16 runs, documented 16-bit A layout) is reused across 4 WMMAs per
// K-step.  Activations are staged to LDS transposed so each lane's B fragment
// is two contiguous ds_load_b128s.  Epilogue stores bf16 output and emits
// per-block per-channel sum/sumsq partials (deterministic, no atomics) so the
// BN stats pass never has to re-read the 67MB raw tensor from HBM.
// ---------------------------------------------------------------------------
template <int TAPS>
__global__ __launch_bounds__(256) void wmma_gemm_kernel(
    const __bf16* __restrict__ W,     // [TAPS][128][K] bf16
    const __bf16* __restrict__ X,     // [B][K][L]      bf16
    __bf16* __restrict__ Yraw,        // [B][128][L]    bf16 raw output
    float* __restrict__ Psum,         // [NPART][128] partial sums
    float* __restrict__ Psq,          // [NPART][128] partial sumsqs
    int K)
{
    constexpr int M = 128;
    __shared__ __attribute__((aligned(16))) __bf16 xs[64][48]; // [n][k], 96B rows

    const int tid  = threadIdx.x;
    const int lane = tid & 31;
    const int wv   = tid >> 5;          // wave id -> M tile (0..7)
    const int n    = lane & 15;         // column within 16-wide tile / A row
    const int kh   = lane >> 4;         // K half selector
    const int l0   = blockIdx.x * 64;
    const int b    = blockIdx.y;

    const __bf16* Xb = X + (size_t)b * K * LSEQ;

    v8f acc[NT_PER_BLOCK] = {};
    for (int tap = 0; tap < TAPS; ++tap) {
        const int shift = (TAPS == 3) ? (tap - 1) : 0;
        const __bf16* Wt = W + (size_t)tap * M * K;
        for (int k0 = 0; k0 < K; k0 += 32) {
            __syncthreads();
            // Stage 32(K) x 64(N) tile transposed into xs[n][k].
            #pragma unroll
            for (int e = 0; e < 8; ++e) {
                int idx = tid + 256 * e;
                int kk  = idx >> 6;                 // 0..31
                int nn  = idx & 63;                 // 0..63
                int col = l0 + nn + shift;
                __bf16 val = (__bf16)0.0f;
                if (TAPS == 1 || (col >= 0 && col < LSEQ))
                    val = Xb[(size_t)(k0 + kk) * LSEQ + col];
                xs[nn][kk] = val;
            }
            __syncthreads();

            // A fragment (reused by 4 WMMAs): row wv*16+n,
            // K runs [k0+8kh, +8) and [k0+16+8kh, +8)
            const __bf16* wp = Wt + (size_t)(wv * 16 + n) * K + k0 + kh * 8;
            v8bf a0 = *reinterpret_cast<const v8bf*>(wp);
            v8bf a1 = *reinterpret_cast<const v8bf*>(wp + 16);
            v16bf av;
            #pragma unroll
            for (int j = 0; j < 8; ++j) { av[j] = a0[j]; av[j + 8] = a1[j]; }

            #pragma unroll
            for (int nt = 0; nt < NT_PER_BLOCK; ++nt) {
                v8bf b0 = *reinterpret_cast<const v8bf*>(&xs[nt * 16 + n][kh * 8]);
                v8bf b1 = *reinterpret_cast<const v8bf*>(&xs[nt * 16 + n][16 + kh * 8]);
                v16bf bv;
                #pragma unroll
                for (int j = 0; j < 8; ++j) { bv[j] = b0[j]; bv[j + 8] = b1[j]; }
                acc[nt] = __builtin_amdgcn_wmma_f32_16x16x32_bf16(
                    false, av, false, bv, (short)0, acc[nt], false, false);
            }
        }
    }

    // Store bf16 output.  C layout: VGPR r -> M = r + 8*kh ; N = lane&15.
    __bf16* Yb = Yraw + (size_t)b * M * LSEQ;
    #pragma unroll
    for (int nt = 0; nt < NT_PER_BLOCK; ++nt) {
        #pragma unroll
        for (int r = 0; r < 8; ++r) {
            int o = wv * 16 + r + kh * 8;
            Yb[(size_t)o * LSEQ + l0 + nt * 16 + n] = (__bf16)acc[nt][r];
        }
    }

    // Per-channel partial sum / sumsq over this block's 64 columns.
    // Lanes sharing a channel differ only in n (lane bits 0..3).
    float* ps = Psum + ((size_t)b * NBLK + blockIdx.x) * 128;
    float* pq = Psq  + ((size_t)b * NBLK + blockIdx.x) * 128;
    #pragma unroll
    for (int r = 0; r < 8; ++r) {
        float s = 0.f, s2 = 0.f;
        #pragma unroll
        for (int nt = 0; nt < NT_PER_BLOCK; ++nt) {
            float v = acc[nt][r];
            s += v; s2 += v * v;
        }
        #pragma unroll
        for (int off = 8; off > 0; off >>= 1) {
            s  += __shfl_xor(s, off);
            s2 += __shfl_xor(s2, off);
        }
        if (n == 0) {
            int o = wv * 16 + r + kh * 8;
            ps[o] = s; pq[o] = s2;
        }
    }
}

// ---------------------------------------------------------------------------
// BN stats finalize: reduce NPART partials per channel -> mean, rstd
// ---------------------------------------------------------------------------
__global__ __launch_bounds__(256) void bn_stats_finalize_kernel(
    const float* __restrict__ Psum, const float* __restrict__ Psq,
    float* __restrict__ mean, float* __restrict__ rstd)
{
    const int c = blockIdx.x;
    const int tid = threadIdx.x;
    float s = 0.f, s2 = 0.f;
    for (int i = tid; i < NPART; i += 256) {
        s  += Psum[(size_t)i * 128 + c];
        s2 += Psq[(size_t)i * 128 + c];
    }
    __shared__ float rs[256], rq[256];
    rs[tid] = s; rq[tid] = s2; __syncthreads();
    for (int st = 128; st > 0; st >>= 1) {
        if (tid < st) { rs[tid] += rs[tid + st]; rq[tid] += rq[tid + st]; }
        __syncthreads();
    }
    if (tid == 0) {
        const float N = (float)BATCH * (float)LSEQ;
        float m = rs[0] / N;
        float var = rq[0] / N - m * m;
        mean[c] = m;
        rstd[c] = rsqrtf(var + 1e-5f);
    }
}

// Apply BN (+ optional ReLU) on bf16 raw, write bf16 at channel offset.
__global__ __launch_bounds__(256) void bn_apply_kernel(
    const __bf16* __restrict__ Y, const float* __restrict__ mean,
    const float* __restrict__ rstd, const float* __restrict__ g,
    const float* __restrict__ be, __bf16* __restrict__ out,
    int outC, int chOff, int relu)
{
    const int C = 128;
    int b = blockIdx.x / C, c = blockIdx.x % C;
    float sc = g[c] * rstd[c];
    float sh = be[c] - mean[c] * sc;
    const __bf16* yr = Y + ((size_t)b * C + c) * LSEQ;
    __bf16* orow = out + ((size_t)b * outC + chOff + c) * LSEQ;
    for (int l = threadIdx.x * 8; l < LSEQ; l += 256 * 8) {
        v8bf x = *reinterpret_cast<const v8bf*>(yr + l);
        v8bf o;
        #pragma unroll
        for (int j = 0; j < 8; ++j) {
            float v = (float)x[j] * sc + sh;
            if (relu) v = fmaxf(v, 0.f);
            o[j] = (__bf16)v;
        }
        *reinterpret_cast<v8bf*>(orow + l) = o;
    }
}

// mean over L of BN(Y) computed analytically from the raw row mean.
__global__ __launch_bounds__(256) void bn_rowmean_kernel(
    const __bf16* __restrict__ Y, const float* __restrict__ mean,
    const float* __restrict__ rstd, const float* __restrict__ g,
    const float* __restrict__ be, float* __restrict__ feat, int featOff)
{
    const int C = 128;
    int b = blockIdx.x / C, c = blockIdx.x % C;
    const __bf16* yr = Y + ((size_t)b * C + c) * LSEQ;
    float s = 0.f;
    for (int l = threadIdx.x * 8; l < LSEQ; l += 256 * 8) {
        v8bf x = *reinterpret_cast<const v8bf*>(yr + l);
        #pragma unroll
        for (int j = 0; j < 8; ++j) s += (float)x[j];
    }
    __shared__ float red[256];
    red[threadIdx.x] = s; __syncthreads();
    for (int st = 128; st > 0; st >>= 1) {
        if (threadIdx.x < st) red[threadIdx.x] += red[threadIdx.x + st];
        __syncthreads();
    }
    if (threadIdx.x == 0) {
        float rm = red[0] / (float)LSEQ;
        float sc = g[c] * rstd[c];
        feat[(size_t)b * 256 + featOff + c] = (rm - mean[c]) * sc + be[c];
    }
}

// Convert input row to bf16 + per-row sum / sumsq (for recon loss).
__global__ __launch_bounds__(256) void prep_input_kernel(
    const float* __restrict__ x, __bf16* __restrict__ xb,
    float* __restrict__ rsum, float* __restrict__ rsumsq)
{
    int row = blockIdx.x;                       // b*CIN + c
    const float* xr = x + (size_t)row * LSEQ;
    __bf16* xo = xb + (size_t)row * LSEQ;
    float s = 0.f, s2 = 0.f;
    for (int l = threadIdx.x; l < LSEQ; l += 256) {
        float v = xr[l];
        xo[l] = (__bf16)v;
        s += v; s2 += v * v;
    }
    __shared__ float rs[256], rq[256];
    rs[threadIdx.x] = s; rq[threadIdx.x] = s2; __syncthreads();
    for (int st = 128; st > 0; st >>= 1) {
        if (threadIdx.x < st) { rs[threadIdx.x] += rs[threadIdx.x + st];
                                rq[threadIdx.x] += rq[threadIdx.x + st]; }
        __syncthreads();
    }
    if (threadIdx.x == 0) { rsum[row] = rs[0]; rsumsq[row] = rq[0]; }
}

// 2048-point radix-2 FFT per (b,c) row in LDS; bf16 real/imag outputs.
__global__ __launch_bounds__(256) void fft2048_kernel(
    const float* __restrict__ x, __bf16* __restrict__ outR, __bf16* __restrict__ outI)
{
    __shared__ float re[LSEQ], im[LSEQ];
    const size_t base = (size_t)blockIdx.x * LSEQ;
    const int tid = threadIdx.x;
    for (int i = tid; i < LSEQ; i += 256) {
        unsigned r = __brev((unsigned)i) >> 21;       // 11-bit reversal
        re[r] = x[base + i];
        im[r] = 0.f;
    }
    __syncthreads();
    for (int s = 1; s <= 11; ++s) {
        int m = 1 << s, h = m >> 1;
        for (int j = tid; j < 1024; j += 256) {
            int grp = j / h, pos = j - grp * h;
            int i0 = grp * m + pos, i1 = i0 + h;
            float ang = -6.283185307179586f * (float)pos / (float)m;
            float sn, cs;
            __sincosf(ang, &sn, &cs);
            float xr = re[i1], xi = im[i1];
            float tr = cs * xr - sn * xi;
            float ti = cs * xi + sn * xr;
            re[i1] = re[i0] - tr; im[i1] = im[i0] - ti;
            re[i0] += tr;         im[i0] += ti;
        }
        __syncthreads();
    }
    for (int i = tid; i < LSEQ; i += 256) {
        outR[base + i] = (__bf16)re[i];
        outI[base + i] = (__bf16)im[i];
    }
}

// f32 -> bf16 weight copy / 3-tap permute (O,I,T) -> (T,O,I)
__global__ void f2bf_kernel(const float* __restrict__ w, __bf16* __restrict__ o, int n)
{
    int i = blockIdx.x * 256 + threadIdx.x;
    if (i < n) o[i] = (__bf16)w[i];
}
__global__ void perm3_kernel(const float* __restrict__ w, __bf16* __restrict__ o, int O, int I)
{
    int i = blockIdx.x * 256 + threadIdx.x;
    int n = O * I * 3;
    if (i < n) {
        int t = i % 3; int rest = i / 3; int ii = rest % I; int oo = rest / I;
        o[((size_t)t * O + oo) * I + ii] = (__bf16)w[i];
    }
}

// ---------------------------------------------------------------------------
// Decoder + reconstruction loss (single workgroup; adds into slot[0])
// ---------------------------------------------------------------------------
__global__ __launch_bounds__(256) void decode_recon_kernel(
    const float* __restrict__ feat, const float* __restrict__ dp_w,
    const float* __restrict__ dp_g, const float* __restrict__ dp_be,
    const float* __restrict__ dr_w, const float* __restrict__ dr_g,
    const float* __restrict__ dr_be, const float* __restrict__ rsum,
    const float* __restrict__ rsumsq, float* __restrict__ y1,
    float* __restrict__ y2, float* __restrict__ slot)
{
    const int Bn = BATCH;
    int tid = threadIdx.x;
    __shared__ float sc1[128], sh1[128], sc2[32], sh2[32], red[256];

    for (int idx = tid; idx < 128 * 128; idx += 256) {      // y1 = dp_w @ f
        int b = idx >> 7, o = idx & 127;
        const float* fr = feat + (size_t)b * 256;
        const float* wr = dp_w + (size_t)o * 256;
        float s = 0.f;
        for (int k = 0; k < 256; ++k) s += wr[k] * fr[k];
        y1[idx] = s;
    }
    __threadfence_block();
    __syncthreads();
    if (tid < 128) {                                        // BN over batch
        float s = 0.f, s2 = 0.f;
        for (int b = 0; b < Bn; ++b) { float v = y1[b * 128 + tid]; s += v; s2 += v * v; }
        float m = s / Bn, var = s2 / Bn - m * m;
        float sc = dp_g[tid] * rsqrtf(var + 1e-5f);
        sc1[tid] = sc; sh1[tid] = dp_be[tid] - m * sc;
    }
    __syncthreads();
    for (int idx = tid; idx < 128 * 32; idx += 256) {       // y2 = dr_w @ relu(bn(y1))
        int b = idx >> 5, c = idx & 31;
        const float* wr = dr_w + (size_t)c * 128;
        float s = 0.f;
        for (int o = 0; o < 128; ++o) {
            float v = fmaxf(y1[b * 128 + o] * sc1[o] + sh1[o], 0.f);
            s += wr[o] * v;
        }
        y2[idx] = s;
    }
    __threadfence_block();
    __syncthreads();
    if (tid < 32) {
        float s = 0.f, s2 = 0.f;
        for (int b = 0; b < Bn; ++b) { float v = y2[b * 32 + tid]; s += v; s2 += v * v; }
        float m = s / Bn, var = s2 / Bn - m * m;
        float sc = dr_g[tid] * rsqrtf(var + 1e-5f);
        sc2[tid] = sc; sh2[tid] = dr_be[tid] - m * sc;
    }
    __syncthreads();
    float part = 0.f;                                       // mean((sr - x)^2)
    for (int idx = tid; idx < 128 * 32; idx += 256) {
        int c = idx & 31;
        float sr = y2[idx] * sc2[c] + sh2[c];
        part += (float)LSEQ * sr * sr - 2.f * sr * rsum[idx] + rsumsq[idx];
    }
    red[tid] = part; __syncthreads();
    for (int st = 128; st > 0; st >>= 1) {
        if (tid < st) red[tid] += red[tid + st];
        __syncthreads();
    }
    if (tid == 0) slot[0] += red[0] / ((float)BATCH * CIN * LSEQ);
}

// ---------------------------------------------------------------------------
// Sinkhorn OT loss (single workgroup, 128 threads; K/C in global scratch)
// ---------------------------------------------------------------------------
__global__ __launch_bounds__(128) void sinkhorn_kernel(
    const float* __restrict__ sf, const float* __restrict__ tf,
    float* __restrict__ Cg, float* __restrict__ Kg, float* __restrict__ slot)
{
    const int n = 128;
    int j = threadIdx.x;
    __shared__ float u[128], v[128], red[128];
    __shared__ float meanC;

    float colsum = 0.f;
    const float* tj = tf + (size_t)j * 256;
    for (int i = 0; i < n; ++i) {
        const float* si = sf + (size_t)i * 256;
        float sq = 0.f;
        for (int k = 0; k < 256; ++k) { float d = si[k] - tj[k]; sq += d * d; }
        float c = sqrtf(fmaxf(sq, 0.f));
        Cg[i * n + j] = c;
        colsum += c;
    }
    red[j] = colsum; __syncthreads();
    for (int st = 64; st > 0; st >>= 1) { if (j < st) red[j] += red[j + st]; __syncthreads(); }
    if (j == 0) meanC = red[0] / (float)(n * n);
    __syncthreads();
    float invm = 1.f / meanC;
    for (int i = 0; i < n; ++i) {
        float c = Cg[i * n + j] * invm;
        Cg[i * n + j] = c;
        Kg[i * n + j] = expf(-c / 0.01f);
    }
    v[j] = 0.f;
    __threadfence_block();
    __syncthreads();
    for (int it = 0; it < 100; ++it) {
        float kv = 0.f;                                  // (K v), thread = row
        for (int jj = 0; jj < n; ++jj) kv += Kg[j * n + jj] * v[jj];
        u[j] = (1.f / 128.f) / (kv + 1e-8f);
        __syncthreads();
        float ku = 0.f;                                  // (K^T u), thread = col
        for (int i = 0; i < n; ++i) ku += Kg[i * n + j] * u[i];
        v[j] = (1.f / 128.f) / (ku + 1e-8f);
        __syncthreads();
    }
    float s = 0.f;
    for (int i = 0; i < n; ++i) s += u[i] * Kg[i * n + j] * v[j] * Cg[i * n + j];
    red[j] = s; __syncthreads();
    for (int st = 64; st > 0; st >>= 1) { if (j < st) red[j] += red[j + st]; __syncthreads(); }
    if (j == 0) slot[0] = red[0] / (float)(n * n);
}

// ---------------------------------------------------------------------------
// Contrastive loss (single workgroup, thread per row, online logsumexp)
// ---------------------------------------------------------------------------
__global__ __launch_bounds__(128) void contrastive_kernel(
    const float* __restrict__ sf, const float* __restrict__ tf, float* __restrict__ slot)
{
    int i = threadIdx.x;
    __shared__ float ns[128], nt[128], red[128];
    const float* si = sf + (size_t)i * 256;
    const float* ti = tf + (size_t)i * 256;
    float s2 = 0.f, t2 = 0.f;
    for (int k = 0; k < 256; ++k) { s2 += si[k] * si[k]; t2 += ti[k] * ti[k]; }
    ns[i] = sqrtf(s2); nt[i] = sqrtf(t2);
    __syncthreads();
    float pos = 0.f, m = -1e30f, ssum = 0.f;
    for (int j = 0; j < 128; ++j) {
        const float* tj = tf + (size_t)j * 256;
        float d = 0.f;
        for (int k = 0; k < 256; ++k) d += si[k] * tj[k];
        float sim = d / (ns[i] * nt[j]);
        float val = sim;
        if (j == i) { pos = sim; val = 0.f; }   // neg diag zeroed
        if (val > m) { ssum = ssum * expf(m - val) + 1.f; m = val; }
        else         { ssum += expf(val - m); }
    }
    float loss = (m + logf(ssum)) - pos;
    red[i] = loss; __syncthreads();
    for (int st = 64; st > 0; st >>= 1) { if (i < st) red[i] += red[i + st]; __syncthreads(); }
    if (i == 0) slot[0] = red[0] / 128.f;
}

// Classifier NLL (single workgroup, thread per sample)
__global__ __launch_bounds__(128) void cls_kernel(
    const float* __restrict__ sf, const int* __restrict__ y,
    const float* __restrict__ cw, const float* __restrict__ cb, float* __restrict__ slot)
{
    int b = threadIdx.x;
    __shared__ float red[128];
    const float* fb = sf + (size_t)b * 256;
    float logits[NCLS];
    float m = -1e30f;
    for (int c = 0; c < NCLS; ++c) {
        const float* w = cw + (size_t)c * 256;
        float s = cb[c];
        for (int k = 0; k < 256; ++k) s += w[k] * fb[k];
        logits[c] = s;
        m = fmaxf(m, s);
    }
    float se = 0.f;
    for (int c = 0; c < NCLS; ++c) se += expf(logits[c] - m);
    float loss = (m + logf(se)) - logits[y[b]];
    red[b] = loss; __syncthreads();
    for (int st = 64; st > 0; st >>= 1) { if (b < st) red[b] += red[b + st]; __syncthreads(); }
    if (b == 0) slot[0] = red[0] / 128.f;
}

__global__ void init_slots_kernel(float* slots)
{
    if (threadIdx.x < 8) slots[threadIdx.x] = 0.f;
}
__global__ void combine_kernel(const float* __restrict__ slots, float* __restrict__ out)
{
    out[0] = slots[0] + slots[1] + slots[2] + slots[3];
}

// ---------------------------------------------------------------------------
// Orchestration
// ---------------------------------------------------------------------------
extern "C" void kernel_launch(void* const* d_in, const int* in_sizes, int n_in,
                              void* d_out, int out_size, void* d_ws, size_t ws_size,
                              hipStream_t stream)
{
    (void)in_sizes; (void)n_in; (void)out_size; (void)ws_size;

    const float* src_x = (const float*)d_in[0];
    const int*   src_y = (const int*)d_in[1];
    const float* trg_x = (const float*)d_in[2];
    const float* w1     = (const float*)d_in[3];
    const float* w2     = (const float*)d_in[4];
    const float* bn_r_g = (const float*)d_in[5];
    const float* bn_r_b = (const float*)d_in[6];
    const float* bn_p_g = (const float*)d_in[7];
    const float* bn_p_b = (const float*)d_in[8];
    const float* pr_w   = (const float*)d_in[9];
    const float* pr_g   = (const float*)d_in[11];
    const float* pr_be  = (const float*)d_in[12];
    const float* pp_w   = (const float*)d_in[13];
    const float* pp_g   = (const float*)d_in[15];
    const float* pp_be  = (const float*)d_in[16];
    const float* fp_w   = (const float*)d_in[17];
    const float* fp_g   = (const float*)d_in[19];
    const float* fp_be  = (const float*)d_in[20];
    const float* t1_w   = (const float*)d_in[21];
    const float* t1_g   = (const float*)d_in[23];
    const float* t1_be  = (const float*)d_in[24];
    const float* t2_w   = (const float*)d_in[25];
    const float* t2_g   = (const float*)d_in[27];
    const float* t2_be  = (const float*)d_in[28];
    const float* dp_w   = (const float*)d_in[29];
    const float* dp_g   = (const float*)d_in[31];
    const float* dp_be  = (const float*)d_in[32];
    const float* dr_w   = (const float*)d_in[33];
    const float* dr_g   = (const float*)d_in[35];
    const float* dr_be  = (const float*)d_in[36];
    const float* cls_w  = (const float*)d_in[37];
    const float* cls_b  = (const float*)d_in[38];

    char* ws = (char*)d_ws;
    size_t off = 0;
    auto alloc = [&](size_t bytes) -> void* {
        void* p = ws + off;
        off = (off + bytes + 255) & ~(size_t)255;
        return p;
    };

    const size_t catBF = (size_t)BATCH * 256 * LSEQ * sizeof(__bf16);  // 134 MB
    const size_t midBF = (size_t)BATCH * 128 * LSEQ * sizeof(__bf16);  //  67 MB
    const size_t inBF  = (size_t)BATCH * CIN * LSEQ * sizeof(__bf16);  //  17 MB

    __bf16* Yraw = (__bf16*)alloc(midBF);      // raw GEMM output (bf16)
    __bf16* CAT  = (__bf16*)alloc(catBF);
    __bf16* B0   = (__bf16*)alloc(midBF);
    __bf16* XB   = (__bf16*)alloc(inBF);
    __bf16* FR   = (__bf16*)alloc(inBF);
    __bf16* FI   = (__bf16*)alloc(inBF);

    __bf16* w1b = (__bf16*)alloc(128 * 32 * 2);
    __bf16* w2b = (__bf16*)alloc(128 * 32 * 2);
    __bf16* prb = (__bf16*)alloc(128 * 128 * 2);
    __bf16* ppb = (__bf16*)alloc(128 * 128 * 2);
    __bf16* fpb = (__bf16*)alloc(128 * 256 * 2);
    __bf16* t1b = (__bf16*)alloc(3 * 128 * 32 * 2);
    __bf16* t2b = (__bf16*)alloc(3 * 128 * 128 * 2);

    float* Psum   = (float*)alloc((size_t)NPART * 128 * 4);
    float* Psq    = (float*)alloc((size_t)NPART * 128 * 4);
    float* mean   = (float*)alloc(128 * 4);
    float* rstd   = (float*)alloc(128 * 4);
    float* sfeat  = (float*)alloc(BATCH * 256 * 4);
    float* tfeat  = (float*)alloc(BATCH * 256 * 4);
    float* rsum_s = (float*)alloc(BATCH * CIN * 4);
    float* rsq_s  = (float*)alloc(BATCH * CIN * 4);
    float* rsum_t = (float*)alloc(BATCH * CIN * 4);
    float* rsq_t  = (float*)alloc(BATCH * CIN * 4);
    float* dec_y1 = (float*)alloc(128 * 128 * 4);
    float* dec_y2 = (float*)alloc(128 * 32 * 4);
    float* Cg     = (float*)alloc(128 * 128 * 4);
    float* Kg     = (float*)alloc(128 * 128 * 4);
    float* slots  = (float*)alloc(16 * 4);

    init_slots_kernel<<<1, 32, 0, stream>>>(slots);

    // Weight conversion to bf16 (conv biases cancel inside the following BN)
    f2bf_kernel<<<16, 256, 0, stream>>>(w1, w1b, 128 * 32);
    f2bf_kernel<<<16, 256, 0, stream>>>(w2, w2b, 128 * 32);
    f2bf_kernel<<<64, 256, 0, stream>>>(pr_w, prb, 128 * 128);
    f2bf_kernel<<<64, 256, 0, stream>>>(pp_w, ppb, 128 * 128);
    f2bf_kernel<<<128, 256, 0, stream>>>(fp_w, fpb, 128 * 256);
    perm3_kernel<<<48, 256, 0, stream>>>(t1_w, t1b, 128, 32);
    perm3_kernel<<<192, 256, 0, stream>>>(t2_w, t2b, 128, 128);

    const dim3 gemmGrid(NBLK, BATCH);
    const int  rows = BATCH * CIN;

    auto encode = [&](const float* x, float* feat, float* rsum, float* rsq) {
        prep_input_kernel<<<rows, 256, 0, stream>>>(x, XB, rsum, rsq);
        fft2048_kernel<<<rows, 256, 0, stream>>>(x, FR, FI);

        // real path: w1 einsum -> bn_r -> pr conv1 -> bn -> CAT[0:128]
        wmma_gemm_kernel<1><<<gemmGrid, 256, 0, stream>>>(w1b, FR, Yraw, Psum, Psq, 32);
        bn_stats_finalize_kernel<<<128, 256, 0, stream>>>(Psum, Psq, mean, rstd);
        bn_apply_kernel<<<BATCH * 128, 256, 0, stream>>>(Yraw, mean, rstd, bn_r_g, bn_r_b, B0, 128, 0, 0);
        wmma_gemm_kernel<1><<<gemmGrid, 256, 0, stream>>>(prb, B0, Yraw, Psum, Psq, 128);
        bn_stats_finalize_kernel<<<128, 256, 0, stream>>>(Psum, Psq, mean, rstd);
        bn_apply_kernel<<<BATCH * 128, 256, 0, stream>>>(Yraw, mean, rstd, pr_g, pr_be, CAT, 256, 0, 0);

        // imag path: w2 einsum -> bn_p -> pp conv1 -> bn -> CAT[128:256]
        wmma_gemm_kernel<1><<<gemmGrid, 256, 0, stream>>>(w2b, FI, Yraw, Psum, Psq, 32);
        bn_stats_finalize_kernel<<<128, 256, 0, stream>>>(Psum, Psq, mean, rstd);
        bn_apply_kernel<<<BATCH * 128, 256, 0, stream>>>(Yraw, mean, rstd, bn_p_g, bn_p_b, B0, 128, 0, 0);
        wmma_gemm_kernel<1><<<gemmGrid, 256, 0, stream>>>(ppb, B0, Yraw, Psum, Psq, 128);
        bn_stats_finalize_kernel<<<128, 256, 0, stream>>>(Psum, Psq, mean, rstd);
        bn_apply_kernel<<<BATCH * 128, 256, 0, stream>>>(Yraw, mean, rstd, pp_g, pp_be, CAT, 256, 128, 0);

        // freq fuse: fp conv1 (K=256) -> bn -> mean over L -> feat[:, :128]
        wmma_gemm_kernel<1><<<gemmGrid, 256, 0, stream>>>(fpb, CAT, Yraw, Psum, Psq, 256);
        bn_stats_finalize_kernel<<<128, 256, 0, stream>>>(Psum, Psq, mean, rstd);
        bn_rowmean_kernel<<<BATCH * 128, 256, 0, stream>>>(Yraw, mean, rstd, fp_g, fp_be, feat, 0);

        // time path: t1 conv3 -> bn+relu -> t2 conv3 -> bn -> mean -> feat[:, 128:]
        wmma_gemm_kernel<3><<<gemmGrid, 256, 0, stream>>>(t1b, XB, Yraw, Psum, Psq, 32);
        bn_stats_finalize_kernel<<<128, 256, 0, stream>>>(Psum, Psq, mean, rstd);
        bn_apply_kernel<<<BATCH * 128, 256, 0, stream>>>(Yraw, mean, rstd, t1_g, t1_be, B0, 128, 0, 1);
        wmma_gemm_kernel<3><<<gemmGrid, 256, 0, stream>>>(t2b, B0, Yraw, Psum, Psq, 128);
        bn_stats_finalize_kernel<<<128, 256, 0, stream>>>(Psum, Psq, mean, rstd);
        bn_rowmean_kernel<<<BATCH * 128, 256, 0, stream>>>(Yraw, mean, rstd, t2_g, t2_be, feat, 128);
    };

    encode(src_x, sfeat, rsum_s, rsq_s);
    encode(trg_x, tfeat, rsum_t, rsq_t);

    decode_recon_kernel<<<1, 256, 0, stream>>>(sfeat, dp_w, dp_g, dp_be, dr_w, dr_g, dr_be,
                                               rsum_s, rsq_s, dec_y1, dec_y2, slots + 0);
    decode_recon_kernel<<<1, 256, 0, stream>>>(tfeat, dp_w, dp_g, dp_be, dr_w, dr_g, dr_be,
                                               rsum_t, rsq_t, dec_y1, dec_y2, slots + 0);
    sinkhorn_kernel<<<1, 128, 0, stream>>>(sfeat, tfeat, Cg, Kg, slots + 1);
    contrastive_kernel<<<1, 128, 0, stream>>>(sfeat, tfeat, slots + 2);
    cls_kernel<<<1, 128, 0, stream>>>(sfeat, src_y, cls_w, cls_b, slots + 3);

    combine_kernel<<<1, 1, 0, stream>>>(slots, (float*)d_out);
}
